// MHSA_68161130987931
// MI455X (gfx1250) — matrix operand order repaired and checked by reading, not compiled, not run on hardware
//
#include <hip/hip_runtime.h>

// ---------------------------------------------------------------------------
// MHSA (bottleneck attention) for MI455X / gfx1250.
// B=16, C=64, W=H=64, N=4096.
// Flash-attention fusion, f16 WMMA (v_wmma_f32_16x16x32_f16), online softmax,
// double-buffered GLOBAL_LOAD_ASYNC_TO_LDS_B128 staging (ASYNCcnt pipeline),
// Q fragments resident in VGPRs across the whole key loop.
// S[n,m] = [q;pos](:,n) . [k;q](:,m)  (K=128 channel-concat trick)
// O[c,n] = sum_m V[c,m] * P[n,m]
// ---------------------------------------------------------------------------

typedef __attribute__((ext_vector_type(16))) _Float16 v16h;
typedef __attribute__((ext_vector_type(8)))  _Float16 v8h;
typedef __attribute__((ext_vector_type(8)))  float    v8f;
typedef __attribute__((ext_vector_type(4)))  int      v4i;

union U16 { v16h v; v8h h[2]; };

#define BATCH 16
#define CCH   64
#define NSP   4096

#if defined(__has_builtin)
#  if __has_builtin(__builtin_amdgcn_global_load_async_to_lds_b128) && \
      __has_builtin(__builtin_amdgcn_s_wait_asynccnt)
#    define USE_ASYNC 1
#  else
#    define USE_ASYNC 0
#  endif
#else
#  define USE_ASYNC 0
#endif

// copy 16 bytes global -> LDS (async on gfx1250, sync fallback otherwise)
__device__ __forceinline__ void cp16(_Float16* ldst, const _Float16* gsrc) {
#if USE_ASYNC
    __builtin_amdgcn_global_load_async_to_lds_b128(
        (__attribute__((address_space(1))) v4i*)(gsrc),
        (__attribute__((address_space(3))) v4i*)(ldst),
        0, 0);
#else
    *(v8h*)ldst = *(const v8h*)gsrc;
#endif
}

// ---------------------------------------------------------------------------
// 1) q/k/v = 1x1 conv (channel mixing), stored as f16.
//    q,k : [b][n][c]  (channel-last, feeds A/B rows of the S GEMM)
//    v   : [b][c][m]  (channel-major, feeds A rows of the O GEMM)
// ---------------------------------------------------------------------------
__global__ __launch_bounds__(256) void qkv_kernel(
    const float* __restrict__ x,
    const float* __restrict__ Wq, const float* __restrict__ bq,
    const float* __restrict__ Wk, const float* __restrict__ bk,
    const float* __restrict__ Wv, const float* __restrict__ bv,
    _Float16* __restrict__ qo, _Float16* __restrict__ ko,
    _Float16* __restrict__ vo)
{
    const int n  = blockIdx.x * 256 + threadIdx.x;   // spatial index
    const int cg = blockIdx.y;                       // channel group (16 ch)
    const int b  = blockIdx.z;

    float aq[16], ak[16], av[16];
#pragma unroll
    for (int i = 0; i < 16; ++i) {
        aq[i] = bq[cg*16+i]; ak[i] = bk[cg*16+i]; av[i] = bv[cg*16+i];
    }
    const float* xb = x + (size_t)b*CCH*NSP + n;
    for (int cin = 0; cin < CCH; ++cin) {
        float xv = xb[(size_t)cin*NSP];              // coalesced over n
#pragma unroll
        for (int i = 0; i < 16; ++i) {               // weight idx uniform -> SGPR loads
            int row = cg*16 + i;
            aq[i] = fmaf(Wq[row*CCH+cin], xv, aq[i]);
            ak[i] = fmaf(Wk[row*CCH+cin], xv, ak[i]);
            av[i] = fmaf(Wv[row*CCH+cin], xv, av[i]);
        }
    }
    union { v8h v[2]; _Float16 s[16]; } pq, pk;
#pragma unroll
    for (int i = 0; i < 16; ++i) { pq.s[i] = (_Float16)aq[i]; pk.s[i] = (_Float16)ak[i]; }
    v8h* qdst = (v8h*)(qo + ((size_t)b*NSP + n)*CCH + cg*16);
    v8h* kdst = (v8h*)(ko + ((size_t)b*NSP + n)*CCH + cg*16);
    qdst[0] = pq.v[0]; qdst[1] = pq.v[1];
    kdst[0] = pk.v[0]; kdst[1] = pk.v[1];
#pragma unroll
    for (int i = 0; i < 16; ++i)
        vo[((size_t)b*CCH + cg*16 + i)*NSP + n] = (_Float16)av[i];
}

// ---------------------------------------------------------------------------
// 2) pos[n][c] = rel_h[c][h] + rel_w[c][w],  n = w*H + h
// ---------------------------------------------------------------------------
__global__ __launch_bounds__(256) void pos_kernel(
    const float* __restrict__ rel_h, const float* __restrict__ rel_w,
    _Float16* __restrict__ po)
{
    int idx = blockIdx.x * 256 + threadIdx.x;
    int c = idx & 63;
    int n = idx >> 6;
    int w = n >> 6, h = n & 63;
    po[(size_t)n*CCH + c] = (_Float16)(rel_h[c*64 + h] + rel_w[c*64 + w]);
}

// ---------------------------------------------------------------------------
// 3) Fused flash attention. Block = (b, 64 query rows), 4 waves.
//    Wave w owns query rows [16w,16w+16). Streams 64-wide key tiles with
//    double-buffered async-to-LDS copies; Q A-fragments live in VGPRs.
// ---------------------------------------------------------------------------
__global__ __launch_bounds__(128) void attn_kernel(
    const _Float16* __restrict__ qg,   // [B][N][C]
    const _Float16* __restrict__ kg,   // [B][N][C]
    const _Float16* __restrict__ vg,   // [B][C][N]
    const _Float16* __restrict__ pg,   // [N][C]
    float* __restrict__ out)           // [B][C][N]
{
    __shared__ _Float16 sK[2][64*128]; // rows m: [k(64) | q(64)]  (double buffer)
    __shared__ _Float16 sV[2][64*64];  // [c][m]                   (double buffer)
    __shared__ _Float16 sP[64*64];     // [n_local][m] (per-wave 16-row slice)
    __shared__ float    sStat[64];     // per-row alpha / l broadcast

    const int tid  = threadIdx.x;
    const int wave = tid >> 5;
    const int lane = tid & 31;
    const int r0   = lane & 15;        // N-dim position inside a 16x16 tile
    const int hi   = lane >> 4;        // lane-half selector (ISA layout)
    const int n0   = blockIdx.x * 64;
    const int b    = blockIdx.y;
    const int row  = tid >> 1;         // staging: 64 rows x 2 half-rows
    const int part = tid & 1;

    // ---- stage Q tile ([q|pos]) through sK[0], lift A-fragments to VGPRs ----
    {
        const _Float16* s = part ? (pg + (size_t)(n0 + row)*CCH)
                                 : (qg + ((size_t)b*NSP + n0 + row)*CCH);
        _Float16* d = &sK[0][row*128 + part*64];
#pragma unroll
        for (int i = 0; i < 8; ++i) cp16(d + i*8, s + i*8);
    }
#if USE_ASYNC
    __builtin_amdgcn_s_wait_asynccnt(0);
#endif
    __syncthreads();
    U16 Aq[4];                                      // ISA 16x32 A layout, K=128
#pragma unroll
    for (int kc = 0; kc < 4; ++kc) {
        const int abase = (wave*16 + r0)*128 + kc*32 + hi*8;
        Aq[kc].h[0] = *(const v8h*)(&sK[0][abase]);
        Aq[kc].h[1] = *(const v8h*)(&sK[0][abase + 16]);
    }
    __syncthreads();                                // done with Q scratch in sK[0]

    const v8f zf = {0.f,0.f,0.f,0.f,0.f,0.f,0.f,0.f};
    v8f Oacc[4] = {zf, zf, zf, zf};                 // [c-tile 0..3] of O[c, n=r0]
    float mrow[8], lrow[8];
#pragma unroll
    for (int r = 0; r < 8; ++r) { mrow[r] = -1e30f; lrow[r] = 0.f; }

    // stage K/V tile (12 b128 per thread = 12 ASYNCcnt per wave)
    auto stage_kv = [&](int m0, _Float16* skbuf, _Float16* svbuf) {
        const _Float16* srcK = part ? (qg + ((size_t)b*NSP + m0 + row)*CCH)
                                    : (kg + ((size_t)b*NSP + m0 + row)*CCH);
        _Float16* dstK = skbuf + row*128 + part*64;
#pragma unroll
        for (int i = 0; i < 8; ++i) cp16(dstK + i*8, srcK + i*8);
        const _Float16* srcV = vg + ((size_t)b*CCH + row)*NSP + m0 + part*32;
        _Float16* dstV = svbuf + row*64 + part*32;
#pragma unroll
        for (int i = 0; i < 4; ++i) cp16(dstV + i*8, srcV + i*8);
    };

#if USE_ASYNC
    stage_kv(0, sK[0], sV[0]);                      // prime the pipeline
#endif

    for (int mt = 0; mt < NSP/64; ++mt) {
#if USE_ASYNC
        const int mtn = (mt + 1) & (NSP/64 - 1);    // wraps harmlessly on last iter
        stage_kv(mtn*64, sK[(mt+1)&1], sV[(mt+1)&1]);
        __builtin_amdgcn_s_wait_asynccnt(12);       // in-order => tile mt complete
        __syncthreads();                            // all waves' portions visible
        const _Float16* cK = sK[mt & 1];
        const _Float16* cV = sV[mt & 1];
#else
        __syncthreads();                            // prev readers done
        stage_kv(mt*64, sK[0], sV[0]);
        __syncthreads();
        const _Float16* cK = sK[0];
        const _Float16* cV = sV[0];
#endif

        // ---- S tile: 16 (rows) x 64 (cols), K=128 over [q|pos].[k|q] ----
        v8f Sacc[4] = {zf, zf, zf, zf};
#pragma unroll
        for (int kc = 0; kc < 4; ++kc) {            // K chunks of 32
#pragma unroll
            for (int j = 0; j < 4; ++j) {           // 4 m-subtiles
                U16 bf;                             // B: N=key cols, K in ISA table order
                const int bbase = (j*16 + r0)*128 + kc*32 + hi*8;
                bf.h[0] = *(const v8h*)(cK + bbase);
                bf.h[1] = *(const v8h*)(cK + bbase + 16);
                Sacc[j] = __builtin_amdgcn_wmma_f32_16x16x32_f16(
                    false, Aq[kc].v, false, bf.v, (short)0, Sacc[j], false, false);
            }
        }

        // ---- online softmax over the 64 new columns ----
#pragma unroll
        for (int r = 0; r < 8; ++r) {               // row M = r + 8*hi
            float vmax = fmaxf(fmaxf(Sacc[0][r], Sacc[1][r]),
                               fmaxf(Sacc[2][r], Sacc[3][r]));
#pragma unroll
            for (int off = 1; off < 16; off <<= 1)
                vmax = fmaxf(vmax, __shfl_xor(vmax, off, 32));
            float mnew  = fmaxf(mrow[r], vmax);
            float alpha = __expf(mrow[r] - mnew);
            mrow[r] = mnew;
            float rsum = 0.f;
            const int nl = wave*16 + r + 8*hi;
#pragma unroll
            for (int j = 0; j < 4; ++j) {
                float p = __expf(Sacc[j][r] - mnew);
                rsum += p;
                sP[nl*64 + j*16 + r0] = (_Float16)p;
            }
#pragma unroll
            for (int off = 1; off < 16; off <<= 1)
                rsum += __shfl_xor(rsum, off, 32);
            lrow[r] = lrow[r]*alpha + rsum;
            if (r0 == 0) sStat[nl] = alpha;         // broadcast alpha by row n
        }
        // rescale O accumulator: factor depends only on n = r0 (per-lane)
        float ascale = sStat[wave*16 + r0];
#pragma unroll
        for (int ct = 0; ct < 4; ++ct) Oacc[ct] = Oacc[ct] * ascale;

        // ---- O += V (c x m) * P^T (m x n) ----
#pragma unroll
        for (int kc2 = 0; kc2 < 2; ++kc2) {         // m chunks of 32
            U16 pb;                                 // B: N=n cols, K=m
            const int pbase = (wave*16 + r0)*64 + kc2*32 + hi*8;
            pb.h[0] = *(const v8h*)(sP + pbase);
            pb.h[1] = *(const v8h*)(sP + pbase + 16);
#pragma unroll
            for (int ct = 0; ct < 4; ++ct) {        // A: M=c rows
                U16 va;
                const int vbase = (ct*16 + r0)*64 + kc2*32 + hi*8;
                va.h[0] = *(const v8h*)(cV + vbase);
                va.h[1] = *(const v8h*)(cV + vbase + 16);
                Oacc[ct] = __builtin_amdgcn_wmma_f32_16x16x32_f16(
                    false, va.v, false, pb.v, (short)0, Oacc[ct], false, false);
            }
        }
#if USE_ASYNC
        __syncthreads();    // all done reading buf[mt&1] before it is restaged
#endif
    }

    // ---- normalize by running sum and store out[b][c][n] ----
#pragma unroll
    for (int r = 0; r < 8; ++r)
        if (r0 == 0) sStat[wave*16 + r + 8*hi] = lrow[r];
    float linv = 1.0f / sStat[wave*16 + r0];
    const int ncol = n0 + wave*16 + r0;
#pragma unroll
    for (int ct = 0; ct < 4; ++ct)
#pragma unroll
        for (int rr = 0; rr < 8; ++rr) {
            int c = ct*16 + rr + 8*hi;
            out[((size_t)b*CCH + c)*NSP + ncol] = Oacc[ct][rr] * linv;
        }
}

// ---------------------------------------------------------------------------
extern "C" void kernel_launch(void* const* d_in, const int* in_sizes, int n_in,
                              void* d_out, int out_size, void* d_ws, size_t ws_size,
                              hipStream_t stream) {
    (void)in_sizes; (void)n_in; (void)out_size; (void)ws_size;
    const float* x  = (const float*)d_in[0];
    const float* Wq = (const float*)d_in[1];
    const float* bq = (const float*)d_in[2];
    const float* Wk = (const float*)d_in[3];
    const float* bk = (const float*)d_in[4];
    const float* Wv = (const float*)d_in[5];
    const float* bv = (const float*)d_in[6];
    const float* rh = (const float*)d_in[7];
    const float* rw = (const float*)d_in[8];
    float* out = (float*)d_out;

    char* ws = (char*)d_ws;
    const size_t QB = (size_t)BATCH * NSP * CCH * sizeof(_Float16);  // 8 MB
    _Float16* qg = (_Float16*)(ws);
    _Float16* kg = (_Float16*)(ws + QB);
    _Float16* vg = (_Float16*)(ws + 2*QB);
    _Float16* pg = (_Float16*)(ws + 3*QB);   // + 512 KB

    qkv_kernel<<<dim3(NSP/256, 4, BATCH), 256, 0, stream>>>(
        x, Wq, bq, Wk, bk, Wv, bv, qg, kg, vg);
    pos_kernel<<<dim3((NSP*CCH)/256), 256, 0, stream>>>(rh, rw, pg);
    attn_kernel<<<dim3(NSP/64, BATCH), 128, 0, stream>>>(qg, kg, vg, pg, out);
}
